// BidirectionalRNN_43327630082677
// MI455X (gfx1250) — compile-verified, matrix operand order checked by reading, not run on hardware
//
#include <hip/hip_runtime.h>
#include <hip/hip_bf16.h>
#include <math.h>

typedef __bf16 bf16;
typedef __attribute__((ext_vector_type(8)))  __bf16 v8bf;
typedef __attribute__((ext_vector_type(16))) __bf16 v16bf;
typedef __attribute__((ext_vector_type(8)))  float  v8f;

#define EMB   512
#define HID   512
#define OUTD  512
#define BATCH 32
#define SEQ   512
#define NROWS (BATCH * SEQ)   // 16384 tokens
#define LDS_STRIDE 520        // 512 + 8 pad: rows land on distinct 4-bank groups

// ---------- fragment helpers ----------

// A-fragment (16x32 bf16, MxK) from a row-major row pointer already offset to
// this lane's row and K-base (kt*32 + 8*half). Elements 0..7 = K+0..7,
// elements 8..15 = K+16..23  (ISA 7.12.2 16-bit A layout).
__device__ __forceinline__ v16bf frag_a(const bf16* p) {
  union { v16bf v; v8bf h[2]; } u;
  u.h[0] = *(const v8bf*)(p);
  u.h[1] = *(const v8bf*)(p + 16);
  return u.v;
}

// B-fragment from pre-packed weights: layout [nt][kt][lane][16 elems] contiguous.
__device__ __forceinline__ v16bf frag_b(const bf16* __restrict__ wp, int nt, int kt,
                                        int KT, int lane) {
  const bf16* p = wp + (((size_t)(nt * KT + kt)) * 32 + lane) * 16;
  union { v16bf v; v8bf h[2]; } u;
  u.h[0] = *(const v8bf*)(p);
  u.h[1] = *(const v8bf*)(p + 8);
  return u.v;
}

__device__ __forceinline__ v8f wmma_bf16(v16bf a, v16bf b, v8f c) {
  return __builtin_amdgcn_wmma_f32_16x16x32_bf16(false, a, false, b, (short)0, c,
                                                 false, false);
}

// Cheap tanh: one v_exp_f32 + rcp; clamp avoids inf/inf NaN. bf16 output anyway.
__device__ __forceinline__ float fast_tanh(float x) {
  float ax = fminf(fabsf(x), 12.f);
  float e  = __expf(2.f * ax);
  float r  = (e - 1.f) / (e + 1.f);
  return copysignf(r, x);
}

// ---------- small prep kernels ----------

__global__ __launch_bounds__(256) void f32_to_bf16_kernel(
    const float* __restrict__ src, bf16* __restrict__ dst, int n) {
  int i = blockIdx.x * 256 + threadIdx.x;
  if (i < n) dst[i] = (bf16)src[i];
}

// Pack W (K = KT*32 rows, N = 512 cols, row-major f32) into B-fragment layout.
// lane = N within tile (lanes 16-31 repeat N with the other K half),
// element e -> K = kt*32 + e + (e&8) + 8*(lane>=16).
__global__ __launch_bounds__(256) void pack_w_kernel(
    const float* __restrict__ W, bf16* __restrict__ dst, int KT, int total) {
  int idx = blockIdx.x * 256 + threadIdx.x;
  if (idx >= total) return;
  int e    = idx & 15;
  int lane = (idx >> 4) & 31;
  int rest = idx >> 9;
  int kt   = rest % KT;
  int nt   = rest / KT;
  int N = nt * 16 + (lane & 15);
  int K = kt * 32 + e + (e & 8) + ((lane >> 4) << 3);
  dst[idx] = (bf16)W[(size_t)K * 512 + N];
}

// ---------- phase 1: Z[dir] = X @ Wx[dir] + b[dir]  (rows ordered [t][b]) ----------

__global__ __launch_bounds__(256) void gemm_pre_kernel(
    const bf16* __restrict__ XB,
    const bf16* __restrict__ WXF, const bf16* __restrict__ WXB,
    const float* __restrict__ bf_, const float* __restrict__ bb_,
    float* __restrict__ Zf, float* __restrict__ Zb) {
  const int dir  = blockIdx.y;
  const bf16*  wp   = dir ? WXB : WXF;
  const float* bias = dir ? bb_ : bf_;
  float*       Z    = dir ? Zb  : Zf;

  const int lane = threadIdx.x & 31;
  const int wave = threadIdx.x >> 5;
  const int half = lane >> 4;
  const int l15  = lane & 15;

  const int w   = blockIdx.x * 8 + wave;   // 8192 wave-jobs
  const int mt  = w >> 3;                  // 1024 M-tiles
  const int nt0 = (w & 7) * 4;             // 4 N-tiles per wave

  // A rows: output row r = t*32 + b  ->  X row = b*SEQ + t
  const int rOut = mt * 16 + l15;
  const int t = rOut >> 5, b = rOut & 31;
  const bf16* arow = XB + ((size_t)b * SEQ + t) * EMB;

  v8f acc[4];
  for (int j = 0; j < 4; ++j) {
    float bv = bias[(nt0 + j) * 16 + l15];
    for (int r = 0; r < 8; ++r) acc[j][r] = bv;
  }
  for (int kt = 0; kt < 16; ++kt) {
    v16bf a = frag_a(arow + kt * 32 + 8 * half);
    for (int j = 0; j < 4; ++j)
      acc[j] = wmma_bf16(a, frag_b(wp, nt0 + j, kt, 16, lane), acc[j]);
  }
  for (int j = 0; j < 4; ++j)
    for (int r = 0; r < 8; ++r)
      Z[(size_t)(mt * 16 + r + 8 * half) * HID + (nt0 + j) * 16 + l15] = acc[j][r];
}

// ---------- phase 2: persistent recurrence, one workgroup per direction ----------
// Wh B-fragments are step-invariant: keep all 16 (128 VGPRs) register-resident
// so the serial critical path is only LDS A-frags + WMMA + tanh + barrier.

__global__ __launch_bounds__(1024) void rnn_scan_kernel(
    const float* __restrict__ Zf, const float* __restrict__ Zb,
    const bf16* __restrict__ WHF, const bf16* __restrict__ WHB,
    bf16* __restrict__ Hf, bf16* __restrict__ Hb) {
  __shared__ bf16 hbuf[2][BATCH][LDS_STRIDE];   // ping-pong h, 65 KB

  const int dir = blockIdx.x;
  const float* Z  = dir ? Zb  : Zf;
  const bf16*  wh = dir ? WHB : WHF;
  bf16*        H  = dir ? Hb  : Hf;

  const int lane = threadIdx.x & 31;
  const int wave = threadIdx.x >> 5;   // 32 waves = 32 N-tiles
  const int half = lane >> 4;
  const int l15  = lane & 15;
  const int col  = wave * 16 + l15;

  // Persistent Wh fragments for this wave's column tile.
  v16bf wfrag[16];
#pragma unroll
  for (int kt = 0; kt < 16; ++kt) wfrag[kt] = frag_b(wh, wave, kt, 16, lane);

  // h0 = 0
  bf16* hz = &hbuf[0][0][0];
  for (int i = threadIdx.x; i < BATCH * LDS_STRIDE; i += 1024) hz[i] = (bf16)0.f;
  __syncthreads();

  for (int s = 0; s < SEQ; ++s) {
    const int t   = dir ? (SEQ - 1 - s) : s;
    const int cur = s & 1, nxt = cur ^ 1;

    // Prefetch next step's Z tile into cache (one line per lane covers 32 rows).
    const int t2 = dir ? (t - 1) : (t + 1);
    if (t2 >= 0 && t2 < SEQ)
      __builtin_prefetch(Z + (size_t)t2 * BATCH * HID + (size_t)lane * HID + wave * 16, 0, 0);

    // acc <- Z[t] tile (bias already folded in)
    v8f acc0, acc1;
    const float* zc = Z + (size_t)t * BATCH * HID + col;
#pragma unroll
    for (int r = 0; r < 8; ++r) {
      acc0[r] = zc[(size_t)(r + 8 * half) * HID];
      acc1[r] = zc[(size_t)(16 + r + 8 * half) * HID];
    }

    // acc += h_prev @ Wh[:, wave-tile]   (A from LDS, B from registers)
    const bf16* hprev = &hbuf[cur][0][0];
#pragma unroll
    for (int kt = 0; kt < 16; ++kt) {
      const int ko = kt * 32 + 8 * half;
      v16bf a0 = frag_a(hprev + (size_t)l15 * LDS_STRIDE + ko);
      v16bf a1 = frag_a(hprev + (size_t)(16 + l15) * LDS_STRIDE + ko);
      acc0 = wmma_bf16(a0, wfrag[kt], acc0);
      acc1 = wmma_bf16(a1, wfrag[kt], acc1);
    }

    // h = tanh(acc): write LDS (next step) + global H (for output projection)
    bf16* hcur = &hbuf[nxt][0][0];
    bf16* hout = H + (size_t)t * BATCH * HID;
#pragma unroll
    for (int r = 0; r < 8; ++r) {
      int m0 = r + 8 * half, m1 = 16 + r + 8 * half;
      bf16 c0 = (bf16)fast_tanh(acc0[r]);
      bf16 c1 = (bf16)fast_tanh(acc1[r]);
      hcur[(size_t)m0 * LDS_STRIDE + col] = c0;
      hcur[(size_t)m1 * LDS_STRIDE + col] = c1;
      hout[(size_t)m0 * HID + col] = c0;
      hout[(size_t)m1 * HID + col] = c1;
    }
    __syncthreads();
  }
}

// ---------- phase 3: out = [Hf | Hb] @ Wo + bo, rows [t][b] -> out[b][t][:] ----------

__global__ __launch_bounds__(256) void gemm_out_kernel(
    const bf16* __restrict__ Hf, const bf16* __restrict__ Hb,
    const bf16* __restrict__ WOP, const float* __restrict__ bo_,
    float* __restrict__ out) {
  const int lane = threadIdx.x & 31;
  const int wave = threadIdx.x >> 5;
  const int half = lane >> 4;
  const int l15  = lane & 15;

  const int w   = blockIdx.x * 8 + wave;
  const int mt  = w >> 3;
  const int nt0 = (w & 7) * 4;

  const int row = mt * 16 + l15;            // H rows already [t*32+b]
  const bf16* af = Hf + (size_t)row * HID;
  const bf16* ab = Hb + (size_t)row * HID;

  v8f acc[4];
  for (int j = 0; j < 4; ++j) {
    float bv = bo_[(nt0 + j) * 16 + l15];
    for (int r = 0; r < 8; ++r) acc[j][r] = bv;
  }
  for (int kt = 0; kt < 32; ++kt) {
    const bf16* src = (kt < 16) ? af : ab;
    v16bf a = frag_a(src + (kt & 15) * 32 + 8 * half);
    for (int j = 0; j < 4; ++j)
      acc[j] = wmma_bf16(a, frag_b(WOP, nt0 + j, kt, 32, lane), acc[j]);
  }
  for (int j = 0; j < 4; ++j)
    for (int r = 0; r < 8; ++r) {
      int g = mt * 16 + r + 8 * half;
      int b = g & 31, t = g >> 5;
      out[((size_t)b * SEQ + t) * OUTD + (nt0 + j) * 16 + l15] = acc[j][r];
    }
}

// ---------- launch ----------

extern "C" void kernel_launch(void* const* d_in, const int* in_sizes, int n_in,
                              void* d_out, int out_size, void* d_ws, size_t ws_size,
                              hipStream_t stream) {
  (void)in_sizes; (void)n_in; (void)out_size; (void)ws_size;
  const float* X   = (const float*)d_in[0];
  const float* Wf  = (const float*)d_in[1];
  const float* bf_ = (const float*)d_in[2];
  const float* Wb  = (const float*)d_in[3];
  const float* bb_ = (const float*)d_in[4];
  const float* Wo  = (const float*)d_in[5];
  const float* bo_ = (const float*)d_in[6];
  float* out = (float*)d_out;

  char* ws = (char*)d_ws;
  size_t o = 0;
  bf16* XB  = (bf16*)(ws + o); o += (size_t)NROWS * EMB * sizeof(bf16);
  bf16* WXF = (bf16*)(ws + o); o += (size_t)EMB * HID * sizeof(bf16);
  bf16* WHF = (bf16*)(ws + o); o += (size_t)HID * HID * sizeof(bf16);
  bf16* WXB = (bf16*)(ws + o); o += (size_t)EMB * HID * sizeof(bf16);
  bf16* WHB = (bf16*)(ws + o); o += (size_t)HID * HID * sizeof(bf16);
  bf16* WOP = (bf16*)(ws + o); o += (size_t)2 * HID * OUTD * sizeof(bf16);
  float* Zf = (float*)(ws + o); o += (size_t)NROWS * HID * sizeof(float);
  float* Zb = (float*)(ws + o); o += (size_t)NROWS * HID * sizeof(float);
  bf16* Hf  = (bf16*)(ws + o); o += (size_t)NROWS * HID * sizeof(bf16);
  bf16* Hb  = (bf16*)(ws + o); o += (size_t)NROWS * HID * sizeof(bf16);

  const int nX = NROWS * EMB;
  f32_to_bf16_kernel<<<(nX + 255) / 256, 256, 0, stream>>>(X, XB, nX);

  const int tot16 = 16 * 16384, tot32 = 32 * 16384;
  pack_w_kernel<<<(tot16 + 255) / 256, 256, 0, stream>>>(Wf, WXF, 16, tot16);
  pack_w_kernel<<<(tot16 + 255) / 256, 256, 0, stream>>>(Wf + (size_t)EMB * HID, WHF, 16, tot16);
  pack_w_kernel<<<(tot16 + 255) / 256, 256, 0, stream>>>(Wb, WXB, 16, tot16);
  pack_w_kernel<<<(tot16 + 255) / 256, 256, 0, stream>>>(Wb + (size_t)EMB * HID, WHB, 16, tot16);
  pack_w_kernel<<<(tot32 + 255) / 256, 256, 0, stream>>>(Wo, WOP, 32, tot32);

  gemm_pre_kernel<<<dim3(1024, 2), 256, 0, stream>>>(XB, WXF, WXB, bf_, bb_, Zf, Zb);
  rnn_scan_kernel<<<2, 1024, 0, stream>>>(Zf, Zb, WHF, WHB, Hf, Hb);
  gemm_out_kernel<<<1024, 256, 0, stream>>>(Hf, Hb, WOP, bo_, out);
}
